// Block_11321533792295
// MI455X (gfx1250) — matrix operand order repaired and checked by reading, not compiled, not run on hardware
//
#include <hip/hip_runtime.h>
#include <math.h>

#define DIMX   768
#define INNER  192
#define HIDDEN 3072
#define SEQ    4096
#define NBATCH 8
#define TOKENS (NBATCH * SEQ)
#define SCALE_A 0.25f       /* 16^-0.5 */
#define EPS_LN  1e-6f
#define EPS_NRM 1e-12f

typedef __attribute__((ext_vector_type(16))) __bf16 v16bf;
typedef __attribute__((ext_vector_type(8)))  __bf16 v8bf;
typedef __attribute__((ext_vector_type(8)))  float  v8f;

// ---------------------------------------------------------------------------
// LDS byte-offset of a shared-memory pointer (generic -> AS(3) cast).
// ---------------------------------------------------------------------------
__device__ __forceinline__ unsigned lds_off(const void* p)
{
    return (unsigned)(unsigned long long)
           (__attribute__((address_space(3))) const char*)p;
}

// ---------------------------------------------------------------------------
// A fragment (16-bit 16x32, ISA 7.12.2): lane m=lane&15 reads K [kA..kA+7]
// and [kA+16..kA+23], kA = kbase + 8*(lane>=16). Two 16B LDS loads.
// ---------------------------------------------------------------------------
__device__ __forceinline__ v16bf load_a(const __bf16* __restrict__ aBase,
                                        int lda, int kbase, int lane)
{
    const int m  = lane & 15;
    const int kA = kbase + ((lane & 16) ? 8 : 0);
    const __bf16* ap = aBase + (size_t)m * lda + kA;
    v8bf lo = *(const v8bf*)ap;
    v8bf hi = *(const v8bf*)(ap + 16);
    return __builtin_shufflevector(lo, hi, 0, 1, 2, 3, 4, 5, 6, 7,
                                           8, 9, 10, 11, 12, 13, 14, 15);
}

__device__ __forceinline__ v8f wmma_bf16(v16bf a, v16bf b, v8f c)
{
    return __builtin_amdgcn_wmma_f32_16x16x32_bf16(false, a, false, b,
                                                   (short)0, c, false, false);
}

// ---------------------------------------------------------------------------
// Pack fp32 weight W[K][N] (row-major) into bf16 WMMA B-fragment order:
//   P[(((nt*(K/32) + kt)*32) + lane)*16 + i] =
//       W[(kt*32 + 8*(lane>=16) + i + (i&8))*N + nt*16 + (lane&15)]
// After this, each lane's B fragment is one contiguous 32-byte chunk.
// ---------------------------------------------------------------------------
__global__ void k_pack(const float* __restrict__ W, __bf16* __restrict__ P,
                       int K, int N)
{
    int idx = blockIdx.x * blockDim.x + threadIdx.x;   // one per (nt,kt,lane)
    int nkt = K / 32;
    int total = (N / 16) * nkt * 32;
    if (idx >= total) return;
    int lane = idx & 31;
    int t    = idx >> 5;
    int kt   = t % nkt;
    int nt   = t / nkt;
    int n    = nt * 16 + (lane & 15);
    int kb   = kt * 32 + ((lane & 16) ? 8 : 0);
    __bf16* dst = P + (size_t)idx * 16;
    const float* src = W + (size_t)kb * N + n;
#pragma unroll
    for (int i = 0; i < 16; ++i) dst[i] = (__bf16)src[(size_t)(i + (i & 8)) * N];
}

// ---------------------------------------------------------------------------
// Kernel 1: LN1 + q/k projections (WMMA) + per-token L2 norm + A = (q.w_g)*s
// One block = 16 tokens, 384 threads = 12 waves (12 N-tiles over INNER).
// ---------------------------------------------------------------------------
__global__ __launch_bounds__(384)
void k_ln1_qk(const float* __restrict__ x,
              const float* __restrict__ g1, const float* __restrict__ b1,
              const __bf16* __restrict__ wqP, const float* __restrict__ bq,
              const __bf16* __restrict__ wkP, const float* __restrict__ bk,
              const float* __restrict__ wg,
              float* __restrict__ qout, float* __restrict__ kout,
              float* __restrict__ Aout)
{
    __shared__ alignas(32) __bf16 hsh[16][DIMX + 8];
    __shared__ float  qsh[16][INNER];
    __shared__ float  ksh[16][INNER];
    __shared__ float  red[16][24], red2[16][24];
    __shared__ float  mus[16], rsg[16];
    __shared__ float  ssq_q[16], ssq_k[16], arow[16];

    const int tid  = threadIdx.x;
    const int tok0 = blockIdx.x * 16;
    if (tid < 16) { ssq_q[tid] = 0.f; ssq_k[tid] = 0.f; arow[tid] = 0.f; }

    // --- LayerNorm statistics: 16 rows x 24 threads each ---
    const int rid = tid / 24, l24 = tid % 24;
    {
        const float* xr = x + (size_t)(tok0 + rid) * DIMX;
        float s = 0.f, s2 = 0.f;
        for (int c = l24; c < DIMX; c += 24) { float v = xr[c]; s += v; s2 += v * v; }
        red[rid][l24] = s; red2[rid][l24] = s2;
    }
    __syncthreads();
    if (l24 == 0) {
        float S = 0.f, S2 = 0.f;
        for (int i = 0; i < 24; ++i) { S += red[rid][i]; S2 += red2[rid][i]; }
        float mn  = S / DIMX;
        float var = S2 / DIMX - mn * mn;
        mus[rid] = mn; rsg[rid] = rsqrtf(var + EPS_LN);
    }
    __syncthreads();
    for (int idx = tid; idx < 16 * DIMX; idx += 384) {
        int r = idx / DIMX, c = idx % DIMX;
        float v = (x[(size_t)(tok0 + r) * DIMX + c] - mus[r]) * rsg[r] * g1[c] + b1[c];
        hsh[r][c] = (__bf16)v;
    }
    __syncthreads();

    // --- WMMA: q = h @ wq + bq ; k = h @ wk + bk ---
    const int wave = tid >> 5, lane = tid & 31;
    const int col  = wave * 16 + (lane & 15);
    const int nkt  = DIMX / 32;
    {
        const __bf16* bp = wqP + ((size_t)wave * nkt * 32 + lane) * 16;
        v8f c = {};
        for (int kt = 0; kt < nkt; ++kt) {
            __builtin_prefetch(bp + (size_t)(kt + 4) * 512, 0, 1);
            v16bf a = load_a(&hsh[0][0], DIMX + 8, kt * 32, lane);
            v16bf b = *(const v16bf*)(bp + (size_t)kt * 512);
            c = wmma_bf16(a, b, c);
        }
        float bias = bq[col];
#pragma unroll
        for (int r = 0; r < 8; ++r) {
            int row  = r + ((lane & 16) ? 8 : 0);
            float v  = c[r] + bias;
            qsh[row][col] = v;
            atomicAdd(&ssq_q[row], v * v);
        }
    }
    {
        const __bf16* bp = wkP + ((size_t)wave * nkt * 32 + lane) * 16;
        v8f c = {};
        for (int kt = 0; kt < nkt; ++kt) {
            __builtin_prefetch(bp + (size_t)(kt + 4) * 512, 0, 1);
            v16bf a = load_a(&hsh[0][0], DIMX + 8, kt * 32, lane);
            v16bf b = *(const v16bf*)(bp + (size_t)kt * 512);
            c = wmma_bf16(a, b, c);
        }
        float bias = bk[col];
#pragma unroll
        for (int r = 0; r < 8; ++r) {
            int row  = r + ((lane & 16) ? 8 : 0);
            float v  = c[r] + bias;
            ksh[row][col] = v;
            atomicAdd(&ssq_k[row], v * v);
        }
    }
    __syncthreads();

    // --- L2 normalize, write q/k, accumulate A = q . w_g ---
    for (int idx = tid; idx < 16 * INNER; idx += 384) {
        int r = idx / INNER, c = idx % INNER;
        size_t o  = (size_t)(tok0 + r) * INNER + c;
        float qin = 1.f / fmaxf(sqrtf(ssq_q[r]), EPS_NRM);
        float kin = 1.f / fmaxf(sqrtf(ssq_k[r]), EPS_NRM);
        float qn  = qsh[r][c] * qin;
        qout[o] = qn;
        kout[o] = ksh[r][c] * kin;
        atomicAdd(&arow[r], qn * wg[c]);
    }
    __syncthreads();
    if (tid < 16) Aout[tok0 + tid] = arow[tid] * SCALE_A;
}

// ---------------------------------------------------------------------------
// Kernel 2: per-batch: A <- l2norm(A over seq); G[j] = sum_s A[s]*q[s,j]
// ---------------------------------------------------------------------------
__global__ __launch_bounds__(256)
void k_gred(const float* __restrict__ A, const float* __restrict__ q,
            float* __restrict__ G)
{
    __shared__ float red[256];
    __shared__ float inv;
    const int b = blockIdx.x, tid = threadIdx.x;
    const float* Ab = A + (size_t)b * SEQ;
    float s = 0.f;
    for (int i = tid; i < SEQ; i += 256) { float v = Ab[i]; s += v * v; }
    red[tid] = s; __syncthreads();
    for (int st = 128; st > 0; st >>= 1) {
        if (tid < st) red[tid] += red[tid + st];
        __syncthreads();
    }
    if (tid == 0) inv = 1.f / fmaxf(sqrtf(red[0]), EPS_NRM);
    __syncthreads();
    for (int j = tid; j < INNER; j += 256) {
        const float* qb = q + (size_t)b * SEQ * INNER + j;
        float g = 0.f;
        for (int sI = 0; sI < SEQ; ++sI) g += Ab[sI] * qb[(size_t)sI * INNER];
        G[b * INNER + j] = g * inv;
    }
}

// ---------------------------------------------------------------------------
// Kernel 3: mid = (G*k)@w_proj + b_proj + q ; out = mid@w_final + b_final + x
// k-tile staged into LDS with CDNA5 async global->LDS DMA (ASYNCcnt).
// ---------------------------------------------------------------------------
__global__ __launch_bounds__(384)
void k_proj(const float* __restrict__ x, const float* __restrict__ q,
            const float* __restrict__ kv, const float* __restrict__ G,
            const __bf16* __restrict__ wprojP, const float* __restrict__ bproj,
            const __bf16* __restrict__ wfinP,  const float* __restrict__ bfin,
            float* __restrict__ outb)
{
    __shared__ alignas(32) float  kshf[16][INNER];   // 12 KB, async-filled
    __shared__ alignas(32) __bf16 tsh[16][INNER];
    __shared__ alignas(32) __bf16 midsh[16][INNER];
    const int tid   = threadIdx.x;
    const int tok0  = blockIdx.x * 16;
    const int batch = tok0 / SEQ;
    const float* Gb = G + batch * INNER;

    // async copy of the 16x192 fp32 k tile into LDS (2 x b128 per thread)
    {
        const unsigned long long gbase =
            (unsigned long long)(const void*)(kv + (size_t)tok0 * INNER);
        const unsigned l0 = lds_off(&kshf[0][0]);
#pragma unroll
        for (int j = 0; j < 2; ++j) {
            unsigned e    = (unsigned)(j * 384 + tid) * 4u;  // float index
            unsigned loff = l0 + e * 4u;
            unsigned voff = e * 4u;
            asm volatile("global_load_async_to_lds_b128 %0, %1, %2"
                         :: "v"(loff), "v"(voff), "s"(gbase)
                         : "memory");
        }
        asm volatile("s_wait_asynccnt 0x0" ::: "memory");
    }
    __syncthreads();

    for (int idx = tid; idx < 16 * INNER; idx += 384) {
        int r = idx / INNER, c = idx % INNER;
        tsh[r][c] = (__bf16)(Gb[c] * kshf[r][c]);
    }
    __syncthreads();

    const int wave = tid >> 5, lane = tid & 31;
    {
        const int nkt = INNER / 32;            // 6
        int col = wave * 16 + (lane & 15);
        const __bf16* bp = wprojP + ((size_t)wave * nkt * 32 + lane) * 16;
        v8f c = {};
        for (int kt = 0; kt < nkt; ++kt) {
            v16bf a = load_a(&tsh[0][0], INNER, kt * 32, lane);
            v16bf b = *(const v16bf*)(bp + (size_t)kt * 512);
            c = wmma_bf16(a, b, c);
        }
        float bias = bproj[col];
#pragma unroll
        for (int r = 0; r < 8; ++r) {
            int row = r + ((lane & 16) ? 8 : 0);
            midsh[row][col] =
                (__bf16)(c[r] + bias + q[(size_t)(tok0 + row) * INNER + col]);
        }
    }
    __syncthreads();
#pragma unroll
    for (int i = 0; i < 4; ++i) {
        const int nkt = INNER / 32;            // 6
        int nt  = wave * 4 + i;
        int col = nt * 16 + (lane & 15);
        const __bf16* bp = wfinP + ((size_t)nt * nkt * 32 + lane) * 16;
        v8f c = {};
        for (int kt = 0; kt < nkt; ++kt) {
            __builtin_prefetch(bp + (size_t)(kt + 2) * 512, 0, 1);
            v16bf a = load_a(&midsh[0][0], INNER, kt * 32, lane);
            v16bf b = *(const v16bf*)(bp + (size_t)kt * 512);
            c = wmma_bf16(a, b, c);
        }
        float bias = bfin[col];
#pragma unroll
        for (int r = 0; r < 8; ++r) {
            int row  = r + ((lane & 16) ? 8 : 0);
            size_t o = (size_t)(tok0 + row) * DIMX + col;
            outb[o]  = c[r] + bias + x[o];
        }
    }
}

// ---------------------------------------------------------------------------
// Kernel 4: h = LN2(out); y = out + gelu(h@w1+b1)@w2 + b2
// Hidden dim (3072) processed in 4 chunks of 768 staged bf16 in LDS.
// ---------------------------------------------------------------------------
__global__ __launch_bounds__(384)
void k_mlp(const float* __restrict__ outb,
           const float* __restrict__ g2, const float* __restrict__ bb2,
           const __bf16* __restrict__ w1P, const float* __restrict__ bb1,
           const __bf16* __restrict__ w2P, const float* __restrict__ bfin,
           float* __restrict__ y)
{
    __shared__ alignas(32) __bf16 hsh[16][DIMX + 8];
    __shared__ alignas(32) __bf16 csh[16][DIMX];
    __shared__ float  red[16][24], red2[16][24];
    __shared__ float  mus[16], rsg[16];

    const int tid  = threadIdx.x;
    const int tok0 = blockIdx.x * 16;

    const int rid = tid / 24, l24 = tid % 24;
    {
        const float* xr = outb + (size_t)(tok0 + rid) * DIMX;
        float s = 0.f, s2 = 0.f;
        for (int c = l24; c < DIMX; c += 24) { float v = xr[c]; s += v; s2 += v * v; }
        red[rid][l24] = s; red2[rid][l24] = s2;
    }
    __syncthreads();
    if (l24 == 0) {
        float S = 0.f, S2 = 0.f;
        for (int i = 0; i < 24; ++i) { S += red[rid][i]; S2 += red2[rid][i]; }
        float mn  = S / DIMX;
        float var = S2 / DIMX - mn * mn;
        mus[rid] = mn; rsg[rid] = rsqrtf(var + EPS_LN);
    }
    __syncthreads();
    for (int idx = tid; idx < 16 * DIMX; idx += 384) {
        int r = idx / DIMX, c = idx % DIMX;
        float v = (outb[(size_t)(tok0 + r) * DIMX + c] - mus[r]) * rsg[r] * g2[c] + bb2[c];
        hsh[r][c] = (__bf16)v;
    }
    __syncthreads();

    const int wave = tid >> 5, lane = tid & 31;
    const int nkt1 = DIMX / 32;     // 24 k-tiles per chunk (both GEMMs)
    const int nktW2 = HIDDEN / 32;  // 96 total k-tiles in packed w2
    v8f acc[4] = {};

    for (int ch = 0; ch < 4; ++ch) {
        // GEMM1 chunk: hidden = gelu(h @ w1 + b1), columns [ch*768, ch*768+768)
#pragma unroll
        for (int i = 0; i < 4; ++i) {
            int ntg  = ch * 48 + wave * 4 + i;           // global n-tile in w1
            int nloc = (wave * 4 + i) * 16;              // column inside chunk
            int col  = ntg * 16 + (lane & 15);
            const __bf16* bp = w1P + ((size_t)ntg * nkt1 * 32 + lane) * 16;
            v8f c = {};
            for (int kt = 0; kt < nkt1; ++kt) {
                __builtin_prefetch(bp + (size_t)(kt + 4) * 512, 0, 1);
                v16bf a = load_a(&hsh[0][0], DIMX + 8, kt * 32, lane);
                v16bf b = *(const v16bf*)(bp + (size_t)kt * 512);
                c = wmma_bf16(a, b, c);
            }
            float bias = bb1[col];
#pragma unroll
            for (int r = 0; r < 8; ++r) {
                int row = r + ((lane & 16) ? 8 : 0);
                float v = c[r] + bias;
                v = 0.5f * v * (1.0f + erff(v * 0.70710678118654752f));
                csh[row][nloc + (lane & 15)] = (__bf16)v;
            }
        }
        __syncthreads();
        // GEMM2 partial: acc += hidden_chunk @ w2[chunk rows]
#pragma unroll
        for (int i = 0; i < 4; ++i) {
            int nt = wave * 4 + i;
            const __bf16* bp = w2P + ((size_t)nt * nktW2 * 32 + lane) * 16
                                   + (size_t)(ch * nkt1) * 512;
            for (int kt = 0; kt < nkt1; ++kt) {
                __builtin_prefetch(bp + (size_t)(kt + 4) * 512, 0, 1);
                v16bf a = load_a(&csh[0][0], DIMX, kt * 32, lane);
                v16bf b = *(const v16bf*)(bp + (size_t)kt * 512);
                acc[i] = wmma_bf16(a, b, acc[i]);
            }
        }
        __syncthreads();
    }

#pragma unroll
    for (int i = 0; i < 4; ++i) {
        int col = (wave * 4 + i) * 16 + (lane & 15);
        float bias = bfin[col];
#pragma unroll
        for (int r = 0; r < 8; ++r) {
            int row  = r + ((lane & 16) ? 8 : 0);
            size_t o = (size_t)(tok0 + row) * DIMX + col;
            y[o] = acc[i][r] + bias + outb[o];
        }
    }
}

// ---------------------------------------------------------------------------
extern "C" void kernel_launch(void* const* d_in, const int* in_sizes, int n_in,
                              void* d_out, int out_size, void* d_ws, size_t ws_size,
                              hipStream_t stream)
{
    (void)in_sizes; (void)n_in; (void)out_size; (void)ws_size;
    const float* x       = (const float*)d_in[0];
    const float* ln1_g   = (const float*)d_in[1];
    const float* ln1_b   = (const float*)d_in[2];
    const float* wq      = (const float*)d_in[3];
    const float* bq      = (const float*)d_in[4];
    const float* wk      = (const float*)d_in[5];
    const float* bk      = (const float*)d_in[6];
    const float* w_g     = (const float*)d_in[7];
    const float* w_proj  = (const float*)d_in[8];
    const float* b_proj  = (const float*)d_in[9];
    const float* w_final = (const float*)d_in[10];
    const float* b_final = (const float*)d_in[11];
    const float* ln2_g   = (const float*)d_in[12];
    const float* ln2_b   = (const float*)d_in[13];
    const float* w1      = (const float*)d_in[14];
    const float* b1      = (const float*)d_in[15];
    const float* w2      = (const float*)d_in[16];
    const float* b2      = (const float*)d_in[17];
    float* y = (float*)d_out;

    char* p = (char*)d_ws;
    auto alloc = [&](size_t bytes) {
        char* r = p;
        p += (bytes + 255) & ~(size_t)255;
        return (void*)r;
    };
    float*  qbuf     = (float*)alloc((size_t)TOKENS * INNER * 4);
    float*  kbuf     = (float*)alloc((size_t)TOKENS * INNER * 4);
    float*  Abuf     = (float*)alloc((size_t)TOKENS * 4);
    float*  Gbuf     = (float*)alloc((size_t)NBATCH * INNER * 4);
    float*  outbuf   = (float*)alloc((size_t)TOKENS * DIMX * 4);
    __bf16* wq_pk    = (__bf16*)alloc((size_t)DIMX * INNER * 2);
    __bf16* wk_pk    = (__bf16*)alloc((size_t)DIMX * INNER * 2);
    __bf16* wproj_pk = (__bf16*)alloc((size_t)INNER * INNER * 2);
    __bf16* wfin_pk  = (__bf16*)alloc((size_t)INNER * DIMX * 2);
    __bf16* w1_pk    = (__bf16*)alloc((size_t)DIMX * HIDDEN * 2);
    __bf16* w2_pk    = (__bf16*)alloc((size_t)HIDDEN * DIMX * 2);

    auto pack = [&](const float* s, __bf16* d, int K, int N) {
        int total = (N / 16) * (K / 32) * 32;
        k_pack<<<(total + 255) / 256, 256, 0, stream>>>(s, d, K, N);
    };
    pack(wq,      wq_pk,    DIMX,  INNER);
    pack(wk,      wk_pk,    DIMX,  INNER);
    pack(w_proj,  wproj_pk, INNER, INNER);
    pack(w_final, wfin_pk,  INNER, DIMX);
    pack(w1,      w1_pk,    DIMX,  HIDDEN);
    pack(w2,      w2_pk,    HIDDEN, DIMX);

    const int ntiles = TOKENS / 16;
    k_ln1_qk<<<ntiles, 384, 0, stream>>>(x, ln1_g, ln1_b, wq_pk, bq, wk_pk, bk,
                                         w_g, qbuf, kbuf, Abuf);
    k_gred<<<NBATCH, 256, 0, stream>>>(Abuf, qbuf, Gbuf);
    k_proj<<<ntiles, 384, 0, stream>>>(x, qbuf, kbuf, Gbuf, wproj_pk, b_proj,
                                       wfin_pk, b_final, outbuf);
    k_mlp<<<ntiles, 384, 0, stream>>>(outbuf, ln2_g, ln2_b, w1_pk, b1,
                                      w2_pk, b2, y);
}
